// VGAEEdgeFeat_23029614641355
// MI455X (gfx1250) — matrix operand order repaired and checked by reading, not compiled, not run on hardware
//
#include <hip/hip_runtime.h>

typedef __attribute__((ext_vector_type(16))) _Float16 v16h;
typedef __attribute__((ext_vector_type(8)))  float    v8f;

// Problem constants (match reference)
constexpr int ZD   = 64;
constexpr int EF   = 16;
constexpr int HID  = 128;
constexpr int IN_DIM = 2 * ZD + EF;   // 144
constexpr int KPAD = 160;             // pad K to 5 tiles of 32
constexpr int KT   = KPAD / 32;       // 5 K-tiles
constexpr int NT   = HID / 16;        // 8 N-tiles
constexpr int NFRAG = KT * NT;        // 40 B-fragments

// ---------------------------------------------------------------------------
// Prep: W1 [144 x 128] f32 (row-major, W1[k*128+n]) -> f16 B-fragments in ws.
// Fragment (t, n) holds B tile K=[32t,32t+32) x N=[16n,16n+16).
// Per-lane B layout (wave32, 16-bit B 32x16): lane l -> column N = 16n+(l&15),
// halves i=0..15 -> K = 32t + ((l&16)?16:0) + i  (contiguous K run).
// Storage: ws[( (t*8+n)*32 + lane )*16 + i] -> each lane loads 32 contiguous B.
// ---------------------------------------------------------------------------
__global__ void vgae_prep_w1(const float* __restrict__ W1,
                             _Float16* __restrict__ wsh) {
  int tid = blockIdx.x * blockDim.x + threadIdx.x;
  if (tid >= NFRAG * 32) return;
  int f = tid >> 5, l = tid & 31;
  int t = f >> 3,  n = f & 7;
  int N = n * 16 + (l & 15);
  int kbase = t * 32 + ((l & 16) ? 16 : 0);
  v16h v;
#pragma unroll
  for (int i = 0; i < 16; ++i) {
    int k = kbase + i;
    v[i] = (k < IN_DIM) ? (_Float16)W1[k * HID + N] : (_Float16)0.f;
  }
  *reinterpret_cast<v16h*>(wsh + (size_t)tid * 16) = v;
}

// Load 8 consecutive f32 of edge_input row, starting at ks (multiple of 8),
// convert to f16 into a[base..base+7]. Each 8-run never crosses a segment
// boundary: [0,64) z[src], [64,128) z[dst], [128,144) edge_attr, >=144 pad 0.
__device__ inline void load_run8(const float* __restrict__ zs,
                                 const float* __restrict__ zd,
                                 const float* __restrict__ ar,
                                 int ks, v16h& a, int base) {
  float4 lo, hi;
  if (ks < 128) {
    const float* p = (ks < ZD) ? (zs + ks) : (zd + (ks - ZD));
    lo = *reinterpret_cast<const float4*>(p);
    hi = *reinterpret_cast<const float4*>(p + 4);
  } else if (ks < IN_DIM) {
    const float* p = ar + (ks - 2 * ZD);
    lo = *reinterpret_cast<const float4*>(p);
    hi = *reinterpret_cast<const float4*>(p + 4);
  } else {
    lo = make_float4(0.f, 0.f, 0.f, 0.f);
    hi = lo;
  }
  a[base + 0] = (_Float16)lo.x; a[base + 1] = (_Float16)lo.y;
  a[base + 2] = (_Float16)lo.z; a[base + 3] = (_Float16)lo.w;
  a[base + 4] = (_Float16)hi.x; a[base + 5] = (_Float16)hi.y;
  a[base + 6] = (_Float16)hi.z; a[base + 7] = (_Float16)hi.w;
}

// ---------------------------------------------------------------------------
// Main: one wave32 handles 16 edges. 5 K-tiles x 8 N-tiles = 40 WMMAs/wave.
// Per K-tile: preload ALL 8 B fragments (one load clause, in-order returns),
// then issue the 8 WMMAs so execution overlaps the outstanding loads instead
// of draining loadcnt to 0 before every WMMA.
// A layout (16-bit A 16x32): lane l -> row M = l&15; elements 0..15 -> K =
// 32t + khalf + {0..15} (contiguous, matches B),
// khalf = (l&16)?16:0. C/D layout: lane l -> col N = l&15; VGPR r -> row
// M = r + ((l&16)?8:0).
// ---------------------------------------------------------------------------
__global__ void __launch_bounds__(256)
vgae_decode(const float* __restrict__ z,
            const long long* __restrict__ ei,   // int64 edge_index [2, E]
            const float* __restrict__ ea,
            const float* __restrict__ b1,
            const float* __restrict__ W2,
            const float* __restrict__ b2,
            const _Float16* __restrict__ wsh,
            float* __restrict__ out,
            int E, int nTiles) {
  int wid = (blockIdx.x * blockDim.x + threadIdx.x) >> 5;  // wave-uniform
  if (wid >= nTiles) return;                               // uniform exit
  int l = threadIdx.x & 31;
  int m = l & 15;                        // edge row within tile
  int khalf = (l & 16) ? 16 : 0;

  int e = wid * 16 + m;
  if (e >= E) e = E - 1;                 // tail clamp (stores guarded below)
  long long src = ei[e];
  long long dst = ei[(long long)E + e];
  const float* zs = z + src * ZD;
  const float* zd = z + dst * ZD;
  const float* ar = ea + (long long)e * EF;

  v8f acc[NT] = {};
#pragma unroll
  for (int t = 0; t < KT; ++t) {
    // A fragment for this K-tile (gathered from z / edge_attr, cvt to f16).
    v16h a;
    int k0 = t * 32 + khalf;
    load_run8(zs, zd, ar, k0,      a, 0);
    load_run8(zs, zd, ar, k0 + 8,  a, 8);

    // Preload all 8 B fragments of this K-tile before any WMMA.
    v16h bf[NT];
#pragma unroll
    for (int n = 0; n < NT; ++n) {
      bf[n] = *reinterpret_cast<const v16h*>(
          wsh + ((size_t)(t * NT + n) * 32 + (size_t)l) * 16);
    }
#pragma unroll
    for (int n = 0; n < NT; ++n) {
      acc[n] = __builtin_amdgcn_wmma_f32_16x16x32_f16(
          false, a, false, bf[n], (short)0, acc[n], false, false);
    }
  }

  // Layer 2: bias + ReLU, dot with W2 (this lane owns one column N per tile),
  // then reduce across the 16 lanes of each half-wave (M split at lane 16).
  float partial[8] = {0.f, 0.f, 0.f, 0.f, 0.f, 0.f, 0.f, 0.f};
#pragma unroll
  for (int n = 0; n < NT; ++n) {
    int N = n * 16 + m;
    float bias = b1[N];
    float w2n  = W2[N];
#pragma unroll
    for (int r = 0; r < 8; ++r) {
      float h = acc[n][r] + bias;
      h = h > 0.f ? h : 0.f;
      partial[r] = fmaf(h, w2n, partial[r]);
    }
  }
#pragma unroll
  for (int mask = 1; mask <= 8; mask <<= 1) {
#pragma unroll
    for (int r = 0; r < 8; ++r)
      partial[r] += __shfl_xor(partial[r], mask, 32);
  }

  if (m == 0) {                     // lanes 0 (M=0..7) and 16 (M=8..15) store
    float bb = b2[0];
    int base = wid * 16 + ((l & 16) ? 8 : 0);
#pragma unroll
    for (int r = 0; r < 8; ++r) {
      int idx = base + r;
      if (idx < E)
        out[idx] = 1.f / (1.f + __expf(-(partial[r] + bb)));
    }
  }
}

extern "C" void kernel_launch(void* const* d_in, const int* in_sizes, int n_in,
                              void* d_out, int out_size, void* d_ws, size_t ws_size,
                              hipStream_t stream) {
  const float*     z  = (const float*)d_in[0];
  const long long* ei = (const long long*)d_in[1];
  const float*     ea = (const float*)d_in[2];
  const float*     W1 = (const float*)d_in[3];
  const float*     b1 = (const float*)d_in[4];
  const float*     W2 = (const float*)d_in[5];
  const float*     b2 = (const float*)d_in[6];
  _Float16* wsh = (_Float16*)d_ws;       // 40 KB of f16 B-fragments
  float*    out = (float*)d_out;

  int E = out_size;                      // 1.6M edges
  int nTiles = (E + 15) / 16;

  vgae_prep_w1<<<(NFRAG * 32 + 255) / 256, 256, 0, stream>>>(W1, wsh);

  const int wavesPerBlock = 8;           // 256 threads = 8 wave32
  int blocks = (nTiles + wavesPerBlock - 1) / wavesPerBlock;
  vgae_decode<<<blocks, 256, 0, stream>>>(z, ei, ea, b1, W2, b2, wsh, out,
                                          E, nTiles);
}
